// LearnableSSN_15908558864824
// MI455X (gfx1250) — matrix operand order repaired and checked by reading, not compiled, not run on hardware
//
#include <hip/hip_runtime.h>

typedef __bf16 bf16_t;
typedef __attribute__((ext_vector_type(16))) __bf16 v16bf;
typedef __attribute__((ext_vector_type(8)))  float  v8f;

// Exact vector type the async-LDS builtin expects (per hipcc diagnostic):
typedef int v4i_vs __attribute__((vector_size(16)));
typedef __attribute__((address_space(1))) v4i_vs as1_v4i;
typedef __attribute__((address_space(3))) v4i_vs as3_v4i;

#define NE_   1024
#define NN    2048   // NE + NI (combined state width, also GEMM K)
#define BB    256    // batch
#define DD    256    // patch dim
#define NSTEP 200
#define TACC  150    // accumulate for t >= STEPS - LAST_N

// LDS staging geometry for k_step: KB=64 K-panel, rows padded to 72 bf16
#define KB       64
#define ROWPAD   72                       // 64 + 8 bf16 pad -> 144 B row stride
#define A_BYTES  (64  * ROWPAD * 2)       // 9216  B per A buffer (64 rows)
#define B_BYTES  (128 * ROWPAD * 2)       // 18432 B per B buffer (128 rows)
#define SMEM_SZ  (2 * A_BYTES + 2 * B_BYTES)  // 55296 B, double buffered

// ---------------------------------------------------------------------------
// Async global -> LDS b128 copy (ASYNCcnt path), with asm fallback
// ---------------------------------------------------------------------------
__device__ __forceinline__ void async_ld128(char* smem_base, unsigned ldsOff,
                                            const char* gptr) {
#if __has_builtin(__builtin_amdgcn_global_load_async_to_lds_b128)
    __builtin_amdgcn_global_load_async_to_lds_b128(
        (as1_v4i*)gptr, (as3_v4i*)(smem_base + ldsOff), 0, 0);
#else
    asm volatile("global_load_async_to_lds_b128 %0, %1, off"
                 :: "v"(ldsOff), "v"((unsigned long long)gptr) : "memory");
#endif
}

__device__ __forceinline__ void wait_async0() {
#if __has_builtin(__builtin_amdgcn_s_wait_asynccnt)
    __builtin_amdgcn_s_wait_asynccnt(0);
#else
    asm volatile("s_wait_asynccnt 0x0" ::: "memory");
#endif
}

// ---------------------------------------------------------------------------
// 1) Row-sum gains: g = S / (rowsum(W) + eps), one wave32 per row (4096 rows)
// ---------------------------------------------------------------------------
__global__ __launch_bounds__(256) void k_gains(
    const float* __restrict__ wee, const float* __restrict__ wei,
    const float* __restrict__ wie, const float* __restrict__ wii,
    float* __restrict__ gains)  // [4][1024]
{
    int rid  = blockIdx.x * 8 + (threadIdx.x >> 5);  // 0..4095
    int lane = threadIdx.x & 31;
    int mat  = rid >> 10;
    int row  = rid & 1023;
    const float* w = (mat == 0) ? wee : (mat == 1) ? wei : (mat == 2) ? wie : wii;
    const float  s = (mat == 0) ? 3.75f : (mat == 1) ? 1.86f : (mat == 2) ? 3.13f : 1.09f;
    const float* p = w + (size_t)row * 1024;
    float sum = 0.f;
    for (int k = lane; k < 1024; k += 32) sum += p[k];
    #pragma unroll
    for (int off = 16; off > 0; off >>= 1) sum += __shfl_xor(sum, off, 32);
    if (lane == 0) gains[rid] = s / (sum + 1e-12f);
}

// ---------------------------------------------------------------------------
// 2) Build combined, clamped, rescaled, sign-folded bf16 weight matrix
// ---------------------------------------------------------------------------
__global__ __launch_bounds__(256) void k_buildw(
    const float* __restrict__ wee, const float* __restrict__ wei,
    const float* __restrict__ wie, const float* __restrict__ wii,
    const float* __restrict__ gains, bf16_t* __restrict__ Wb)
{
    int j = blockIdx.x * 256 + threadIdx.x;  // 0..2047
    int i = blockIdx.y;                      // 0..2047
    float v;
    if (i < NE_) {
        if (j < NE_) v =  fminf(wee[(size_t)i * 1024 + j], 0.15f)           * gains[i];
        else         v = -fmaxf(wei[(size_t)i * 1024 + (j - NE_)], 0.0005f) * gains[1024 + i];
    } else {
        int ii = i - NE_;
        if (j < NE_) v =  wie[(size_t)ii * 1024 + j]         * gains[2048 + ii];
        else         v = -wii[(size_t)ii * 1024 + (j - NE_)] * gains[3072 + ii];
    }
    Wb[(size_t)i * NN + j] = (bf16_t)v;
}

// ---------------------------------------------------------------------------
// 3) Feedforward currents: Iff[b][n] = sum_d x[b][d] * Wff[n][d]  (one-time)
// ---------------------------------------------------------------------------
__global__ __launch_bounds__(256) void k_iff(
    const float* __restrict__ x, const float* __restrict__ wffE,
    const float* __restrict__ wffI, float* __restrict__ iff)
{
    int gid = blockIdx.x * 256 + threadIdx.x;  // 0..524287
    int b = gid >> 11;
    int n = gid & (NN - 1);
    const float4* x4 = (const float4*)(x + (size_t)b * DD);
    const float4* w4 = (const float4*)((n < NE_) ? (wffE + (size_t)n * DD)
                                                 : (wffI + (size_t)(n - NE_) * DD));
    float s = 0.f;
    #pragma unroll 4
    for (int d = 0; d < DD / 4; d++) {
        float4 a = x4[d], c = w4[d];
        s += a.x * c.x + a.y * c.y + a.z * c.z + a.w * c.w;
    }
    iff[gid] = s;
}

// ---------------------------------------------------------------------------
// 4) Zero-init state, accumulator, and bf16 ping-pong rate buffers
// ---------------------------------------------------------------------------
__global__ __launch_bounds__(256) void k_zero(
    float* __restrict__ rstate, float* __restrict__ acc,
    bf16_t* __restrict__ rb0, bf16_t* __restrict__ rb1)
{
    int gid = blockIdx.x * 256 + threadIdx.x;
    rstate[gid] = 0.f;
    acc[gid]    = 0.f;
    rb0[gid]    = (bf16_t)0.f;
    rb1[gid]    = (bf16_t)0.f;
}

// ---------------------------------------------------------------------------
// Stage one KB=64 K-panel of A (64 rows of r) and B (128 rows of Wbig) into
// LDS buffer `buf` via the async copy engine. 256 threads issue 6 b128 each.
// ---------------------------------------------------------------------------
__device__ __forceinline__ void stage_tiles(
    char* smem_c, int tid, int Mb0, int Nb0, int kb, int buf,
    const bf16_t* __restrict__ r_in, const bf16_t* __restrict__ W)
{
    const unsigned aBase = (unsigned)(buf * A_BYTES);
    const unsigned bBase = (unsigned)(2 * A_BYTES + buf * B_BYTES);
    #pragma unroll
    for (int q = 0; q < 2; q++) {              // A: 64 rows x 64 cols = 512 chunks
        int c  = q * 256 + tid;
        int m  = c >> 3;
        int kl = (c & 7) * 8;
        async_ld128(smem_c, aBase + (unsigned)((m * ROWPAD + kl) * 2),
                    (const char*)(r_in + (size_t)(Mb0 + m) * NN + kb + kl));
    }
    #pragma unroll
    for (int q = 0; q < 4; q++) {              // B: 128 rows x 64 cols = 1024 chunks
        int c  = q * 256 + tid;
        int n  = c >> 3;
        int kl = (c & 7) * 8;
        async_ld128(smem_c, bBase + (unsigned)((n * ROWPAD + kl) * 2),
                    (const char*)(W + (size_t)(Nb0 + n) * NN + kb + kl));
    }
}

// ---------------------------------------------------------------------------
// 5) One dynamics step: I = Iff + r @ Wbig.T via bf16 WMMA with async-LDS
//    double buffering; then Euler update with power-law i/o.
//    Block tile 64(batch) x 128(neurons); wave32 owns a 32x32 block.
// ---------------------------------------------------------------------------
__global__ __launch_bounds__(256) void k_step(
    const bf16_t* __restrict__ r_in,   // [256][2048] bf16 (previous rates)
    bf16_t*       __restrict__ r_out,  // [256][2048] bf16 (updated rates)
    float*        __restrict__ rstate, // [256][2048] f32 master state
    float*        __restrict__ acc,    // [256][2048] f32 accumulator
    const float*  __restrict__ iff,    // [256][2048]
    const bf16_t* __restrict__ W,      // [2048][2048] combined weights
    int accFlag)
{
    extern __shared__ char smem_c[];

    const int tid  = threadIdx.x;
    const int lane = tid & 31;
    const int wave = tid >> 5;
    const int half = lane >> 4;   // 0: lanes 0-15, 1: lanes 16-31
    const int l16  = lane & 15;

    const int Mb0 = blockIdx.x * 64;            // block batch-row base
    const int Nb0 = blockIdx.y * 128;           // block neuron-col base
    const int MbW = (wave & 1) * 32;            // wave offsets inside block
    const int NbW = (wave >> 1) * 32;

    // Accumulators initialized from Iff so WMMA's A*B+C yields total current I.
    v8f c[2][2];
    #pragma unroll
    for (int i = 0; i < 2; i++)
        #pragma unroll
        for (int j = 0; j < 2; j++)
            #pragma unroll
            for (int v = 0; v < 8; v++)
                c[i][j][v] = iff[(size_t)(Mb0 + MbW + i * 16 + v + 8 * half) * NN
                                 + (Nb0 + NbW + j * 16 + l16)];

    // Per-lane fragment source rows inside the LDS tiles
    const int am0 = MbW + l16;        // A rows for m-tile 0 (tile 1: +16)
    const int bn0 = NbW + l16;        // B rows for n-tile 0 (tile 1: +16)
    const int aOff = half * 8;        // A: lanes<16 K 0..7 & 16..23; >=16: 8..15 & 24..31
    const int bOff = half * 8;        // B: lanes<16 K 0..7 & 16..23; >=16: 8..15 & 24..31

    union Frag { v16bf v; uint4 q[2]; };

    stage_tiles(smem_c, tid, Mb0, Nb0, 0, 0, r_in, W);

    for (int kb = 0; kb < NN; kb += KB) {
        const int buf = (kb >> 6) & 1;
        wait_async0();          // own async copies for `buf` have landed
        __syncthreads();        // ... and everyone else's too
        if (kb + KB < NN)
            stage_tiles(smem_c, tid, Mb0, Nb0, kb + KB, buf ^ 1, r_in, W);

        const char* sa = smem_c + buf * A_BYTES;
        const char* sb = smem_c + 2 * A_BYTES + buf * B_BYTES;

        #pragma unroll
        for (int ks = 0; ks < KB; ks += 32) {
            Frag a0, a1, b0, b1;
            a0.q[0] = *(const uint4*)(sa + ((am0)      * ROWPAD + ks + aOff)      * 2);
            a0.q[1] = *(const uint4*)(sa + ((am0)      * ROWPAD + ks + aOff + 16) * 2);
            a1.q[0] = *(const uint4*)(sa + ((am0 + 16) * ROWPAD + ks + aOff)      * 2);
            a1.q[1] = *(const uint4*)(sa + ((am0 + 16) * ROWPAD + ks + aOff + 16) * 2);
            b0.q[0] = *(const uint4*)(sb + ((bn0)      * ROWPAD + ks + bOff)      * 2);
            b0.q[1] = *(const uint4*)(sb + ((bn0)      * ROWPAD + ks + bOff + 16) * 2);
            b1.q[0] = *(const uint4*)(sb + ((bn0 + 16) * ROWPAD + ks + bOff)      * 2);
            b1.q[1] = *(const uint4*)(sb + ((bn0 + 16) * ROWPAD + ks + bOff + 16) * 2);

            c[0][0] = __builtin_amdgcn_wmma_f32_16x16x32_bf16(false, a0.v, false, b0.v, (short)0, c[0][0], false, false);
            c[0][1] = __builtin_amdgcn_wmma_f32_16x16x32_bf16(false, a0.v, false, b1.v, (short)0, c[0][1], false, false);
            c[1][0] = __builtin_amdgcn_wmma_f32_16x16x32_bf16(false, a1.v, false, b0.v, (short)0, c[1][0], false, false);
            c[1][1] = __builtin_amdgcn_wmma_f32_16x16x32_bf16(false, a1.v, false, b1.v, (short)0, c[1][1], false, false);
        }
        // No trailing barrier needed: this wave's LDS reads are consumed above,
        // and the next writes to `buf` only issue after the next top barrier.
    }

    // Epilogue: r += (dt/tau) * (-r + k*relu(I)^2); each element owned by one lane.
    #pragma unroll
    for (int i = 0; i < 2; i++) {
        #pragma unroll
        for (int j = 0; j < 2; j++) {
            #pragma unroll
            for (int v = 0; v < 8; v++) {
                int m = Mb0 + MbW + i * 16 + v + 8 * half;
                int n = Nb0 + NbW + j * 16 + l16;
                size_t idx = (size_t)m * NN + n;
                float I    = c[i][j][v];
                float t    = I > 0.f ? I : 0.f;
                float rate = 0.04f * t * t;                 // k * relu(I)^2
                float alph = (n < NE_) ? 0.05f : 0.1f;      // dt/tauE, dt/tauI
                float r0   = rstate[idx];
                float rn   = r0 + alph * (rate - r0);
                rstate[idx] = rn;
                r_out[idx]  = (bf16_t)rn;
                if (accFlag) acc[idx] += rn;
            }
        }
    }
}

// ---------------------------------------------------------------------------
// 6) Finalize: out = [rE_bar (256x1024) | rI_bar (256x1024)], mean of last 50
// ---------------------------------------------------------------------------
__global__ __launch_bounds__(256) void k_final(
    const float* __restrict__ acc, float* __restrict__ out)
{
    int gid = blockIdx.x * 256 + threadIdx.x;
    int b = gid >> 11;
    int n = gid & (NN - 1);
    float v = acc[gid] * (1.0f / 50.0f);
    if (n < NE_) out[(size_t)b * NE_ + n] = v;
    else         out[(size_t)BB * NE_ + (size_t)b * NE_ + (n - NE_)] = v;
}

// ---------------------------------------------------------------------------
extern "C" void kernel_launch(void* const* d_in, const int* in_sizes, int n_in,
                              void* d_out, int out_size, void* d_ws, size_t ws_size,
                              hipStream_t stream) {
    (void)in_sizes; (void)n_in; (void)out_size; (void)ws_size;
    const float* x    = (const float*)d_in[0];
    const float* wee  = (const float*)d_in[1];
    const float* wei  = (const float*)d_in[2];
    const float* wie  = (const float*)d_in[3];
    const float* wii  = (const float*)d_in[4];
    const float* wffE = (const float*)d_in[5];
    const float* wffI = (const float*)d_in[6];

    char* ws = (char*)d_ws;
    bf16_t* Wb    = (bf16_t*)(ws + 0);          // 2048*2048*2 = 8 MB
    bf16_t* rb0   = (bf16_t*)(ws + 8388608);    // 1 MB
    bf16_t* rb1   = (bf16_t*)(ws + 9437184);    // 1 MB
    float*  rst   = (float*) (ws + 10485760);   // 2 MB
    float*  acc   = (float*) (ws + 12582912);   // 2 MB
    float*  iff   = (float*) (ws + 14680064);   // 2 MB
    float*  gains = (float*) (ws + 16777216);   // 16 KB

    k_gains <<<512, 256, 0, stream>>>(wee, wei, wie, wii, gains);
    k_buildw<<<dim3(8, 2048), 256, 0, stream>>>(wee, wei, wie, wii, gains, Wb);
    k_iff   <<<2048, 256, 0, stream>>>(x, wffE, wffI, iff);
    k_zero  <<<2048, 256, 0, stream>>>(rst, acc, rb0, rb1);

    bf16_t* rin  = rb0;
    bf16_t* rout = rb1;
    for (int t = 0; t < NSTEP; t++) {
        k_step<<<dim3(4, 16), 256, SMEM_SZ, stream>>>(rin, rout, rst, acc, iff, Wb,
                                                      (t >= TACC) ? 1 : 0);
        bf16_t* tmp = rin; rin = rout; rout = tmp;
    }
    k_final<<<2048, 256, 0, stream>>>(acc, (float*)d_out);
}
